// Hybrid_MMoE_PLE_29205777613118
// MI455X (gfx1250) — matrix operand order repaired and checked by reading, not compiled, not run on hardware
//
#include <hip/hip_runtime.h>
#include <hip/hip_bf16.h>

// ---------------- model dimensions (match reference) ----------------
#define Bc   8192
#define Dc   1024
#define Tc   3
#define Ec   8
#define HMc  2048
#define OMc  512
#define Pc   3
#define Sc   4
#define HCc  1024
#define OCc  256
#define THc  256

typedef __attribute__((ext_vector_type(16))) __bf16 v16bf;
typedef __attribute__((ext_vector_type(8)))  float  v8f;

// fp32 -> bf16 round-to-nearest-even (bit trick; avoids __bf16 arithmetic)
static __device__ __forceinline__ unsigned short f2bf(float f) {
    union { float f; unsigned int u; } c; c.f = f;
    unsigned int u = c.u;
    u += 0x7FFFu + ((u >> 16) & 1u);
    return (unsigned short)(u >> 16);
}

// =====================================================================
// WMMA GEMM:  C[M,N] = A[M,K] (bf16) x W[K,N] (bf16), fp32 accumulate.
// Block = 256 threads = 8 waves. Block tile = 128(M) x 64(N).
// Wave w: N-strip = w&3, M-subtiles (w>>2)*4 .. +3  -> 4 accumulators,
// B fragment reused across the 4 WMMAs. K stepped by 32.
// LDS: A tile 128x32 (8 KiB) + B tile 32x64 transposed [n][k] (4 KiB).
// Requires: M%128==0, N%64==0, K%32==0 (all shapes here comply).
// =====================================================================
__global__ __launch_bounds__(256) void wmma_gemm_bias(
    const unsigned short* __restrict__ Abf,
    const unsigned short* __restrict__ Wbf,
    const float* __restrict__ bias, float* __restrict__ C,
    int M, int N, int K, int relu)
{
    __shared__ unsigned short ldsA[128 * 32];   // [m][k]
    __shared__ unsigned short ldsB[64 * 32];    // transposed [n][k]

    const int tid    = threadIdx.x;
    const int wave   = tid >> 5;
    const int lane   = tid & 31;
    const int half   = lane >> 4;               // 0: lanes 0-15, 1: 16-31
    const int m0     = blockIdx.y * 128;
    const int n0     = blockIdx.x * 64;
    const int nstrip = wave & 3;
    const int mbase  = (wave >> 2) * 4;         // first of 4 M-subtiles
    const int nlocal = nstrip * 16 + (lane & 15);

    v8f acc[4];
    #pragma unroll
    for (int i = 0; i < 4; ++i) acc[i] = (v8f){0.f,0.f,0.f,0.f,0.f,0.f,0.f,0.f};

    for (int k0 = 0; k0 < K; k0 += 32) {
        // ---- prefetch next K-tiles (global_prefetch_b8) ----
        if (k0 + 32 < K) {
            __builtin_prefetch(Abf + (size_t)(m0 + (tid >> 1)) * K + k0 + 32, 0, 1);
            __builtin_prefetch(Wbf + (size_t)(k0 + 32 + (tid >> 3)) * N + n0, 0, 1);
        }
        // ---- stage A tile 128x32 bf16: thread t -> row t>>1, 16 elems ----
        {
            const int row  = tid >> 1;
            const int koff = (tid & 1) * 16;
            const uint4* src = (const uint4*)(Abf + (size_t)(m0 + row) * K + k0 + koff);
            uint4* dst = (uint4*)(ldsA + row * 32 + koff);
            dst[0] = src[0];
            dst[1] = src[1];
        }
        // ---- stage B tile 32x64 transposed: thread t -> k=t>>3, 8 cols ----
        {
            const int k  = tid >> 3;
            const int nb = (tid & 7) * 8;
            const uint4 q = *(const uint4*)(Wbf + (size_t)(k0 + k) * N + n0 + nb);
            const unsigned short* e = (const unsigned short*)&q;
            #pragma unroll
            for (int i = 0; i < 8; ++i)
                ldsB[(nb + i) * 32 + k] = e[i];
        }
        __syncthreads();

        // ---- B fragment: lanes 0-15 K=0..15, 16-31 K=16..31 (contiguous) ----
        union { uint4 q[2]; v16bf v; } bfr;
        {
            const uint4* p = (const uint4*)&ldsB[nlocal * 32 + half * 16];
            bfr.q[0] = p[0];
            bfr.q[1] = p[1];
        }
        // ---- 4 A fragments, 4 WMMAs, B reused ----
        #pragma unroll
        for (int i = 0; i < 4; ++i) {
            union { uint4 q[2]; v16bf v; } afr;
            const int row = (mbase + i) * 16 + (lane & 15);
            // elems 0..15: K = half*16 + 0..15 (contiguous, matches B)
            afr.q[0] = *(const uint4*)&ldsA[row * 32 + half * 16];
            afr.q[1] = *(const uint4*)&ldsA[row * 32 + half * 16 + 8];
            acc[i] = __builtin_amdgcn_wmma_f32_16x16x32_bf16(
                         false, afr.v, false, bfr.v, (short)0, acc[i], false, false);
        }
        __syncthreads();
    }

    // ---- epilogue: ISA C layout (VGPR r: M = r + 8*half, N = lane&15) ----
    const int n = n0 + nlocal;
    const float bv = bias ? bias[n] : 0.0f;
    #pragma unroll
    for (int i = 0; i < 4; ++i) {
        #pragma unroll
        for (int r = 0; r < 8; ++r) {
            const int m = m0 + (mbase + i) * 16 + r + 8 * half;
            float v = acc[i][r] + bv;
            if (relu) v = fmaxf(v, 0.0f);
            C[(size_t)m * N + n] = v;
        }
    }
}

// ---------------- fp32 -> bf16 convert ----------------
__global__ void f32_to_bf16_k(const float* __restrict__ src,
                              unsigned short* __restrict__ dst, size_t n)
{
    size_t i = (size_t)blockIdx.x * 256 + threadIdx.x;
    if (i < n) dst[i] = f2bf(src[i]);
}

// ------- LayerNorm + affine + ReLU, fp32 in -> bf16 out, block per row ------
__global__ __launch_bounds__(256) void ln_affine_relu_bf16_k(
    const float* __restrict__ X, const float* __restrict__ g,
    const float* __restrict__ be, unsigned short* __restrict__ out, int H)
{
    const int row = blockIdx.x;
    const int tid = threadIdx.x;
    const float* xr = X + (size_t)row * H;
    unsigned short* orow = out + (size_t)row * H;

    float s = 0.f, ss = 0.f;
    for (int i = tid; i < H; i += 256) { float v = xr[i]; s += v; ss += v * v; }

    __shared__ float bs[256], bss[256];
    bs[tid] = s; bss[tid] = ss;
    __syncthreads();
    for (int o = 128; o > 0; o >>= 1) {
        if (tid < o) { bs[tid] += bs[tid + o]; bss[tid] += bss[tid + o]; }
        __syncthreads();
    }
    const float mean = bs[0] / H;
    const float var  = bss[0] / H - mean * mean;
    const float rstd = rsqrtf(var + 1e-5f);

    for (int i = tid; i < H; i += 256) {
        float v = (xr[i] - mean) * rstd * g[i] + be[i];
        orow[i] = f2bf(fmaxf(v, 0.0f));
    }
}

// ---------------- gate logits + softmax, one wave32 per (t,b) row -----------
__global__ __launch_bounds__(256) void gate_softmax_k(
    const float* __restrict__ x, size_t xTaskStride,
    const float* __restrict__ gw, const float* __restrict__ gb,
    float* __restrict__ out, int Bn, int D, int E, int T)
{
    const int row  = blockIdx.x * 8 + (threadIdx.x >> 5);
    const int lane = threadIdx.x & 31;
    if (row >= T * Bn) return;
    const int t = row / Bn, b = row % Bn;

    const float* xp = x + (size_t)t * xTaskStride + (size_t)b * D;
    const float* wp = gw + (size_t)t * D * E;

    float acc[16];
    for (int e = 0; e < 16; ++e) acc[e] = 0.f;
    for (int k = lane; k < D; k += 32) {
        const float xv = xp[k];
        const float* wr = wp + (size_t)k * E;
        for (int e = 0; e < E; ++e) acc[e] += xv * wr[e];
    }
    for (int e = 0; e < E; ++e) {
        float v = acc[e];
        for (int off = 16; off > 0; off >>= 1) v += __shfl_xor(v, off, 32);
        acc[e] = v + gb[(size_t)t * E + e];
    }
    float mx = acc[0];
    for (int e = 1; e < E; ++e) mx = fmaxf(mx, acc[e]);
    float sum = 0.f;
    for (int e = 0; e < E; ++e) { acc[e] = __expf(acc[e] - mx); sum += acc[e]; }
    const float inv = 1.0f / sum;
    if (lane == 0)
        for (int e = 0; e < E; ++e) out[(size_t)row * E + e] = acc[e] * inv;
}

// ---------------- dst[t,b,o] += gate[t,b,e] * src[b,o] ----------------------
__global__ void axpy_gate_k(const float* __restrict__ gate, int E, int e,
                            const float* __restrict__ src, float* __restrict__ dst,
                            int T, int Bn, int O)
{
    size_t i = (size_t)blockIdx.x * 256 + threadIdx.x;
    const size_t total = (size_t)T * Bn * O;
    if (i >= total) return;
    const int o = (int)(i % O);
    const size_t bo = i / O;
    const int b = (int)(bo % Bn);
    const int t = (int)(bo / Bn);
    const float gv = gate[((size_t)t * Bn + b) * E + e];
    dst[i] += gv * src[(size_t)b * O + o];
}

// ---------------- out[i] = (a[i] + a[i+st] + a[i+2st]) / 3 ------------------
__global__ void mean3_k(const float* __restrict__ a, float* __restrict__ out,
                        size_t n, size_t st)
{
    size_t i = (size_t)blockIdx.x * 256 + threadIdx.x;
    if (i < n) out[i] = (a[i] + a[i + st] + a[i + 2 * st]) * (1.0f / 3.0f);
}

__global__ void fzero_k(float* __restrict__ p, size_t n)
{
    size_t i = (size_t)blockIdx.x * 256 + threadIdx.x;
    if (i < n) p[i] = 0.0f;
}

// ---------------- tower output: out[b] = dot(th[b,:], w2) + b2 --------------
__global__ void tower_out_k(const float* __restrict__ th,
                            const float* __restrict__ w2,
                            const float* __restrict__ b2,
                            float* __restrict__ out, int Bn, int H)
{
    const int b = blockIdx.x * 256 + threadIdx.x;
    if (b >= Bn) return;
    float s = b2[0];
    const float* r = th + (size_t)b * H;
    for (int i = 0; i < H; ++i) s += r[i] * w2[i];
    out[b] = s;
}

// =====================================================================
// host-side orchestration
// =====================================================================
static inline void cvt(const float* w, unsigned short* dst, size_t n, hipStream_t s) {
    f32_to_bf16_k<<<dim3((unsigned)((n + 255) / 256)), 256, 0, s>>>(w, dst, n);
}
static inline void gemm(const unsigned short* A, const unsigned short* W,
                        const float* bias, float* C,
                        int M, int N, int K, int relu, hipStream_t s) {
    dim3 g(N / 64, M / 128);
    wmma_gemm_bias<<<g, 256, 0, s>>>(A, W, bias, C, M, N, K, relu);
}
static inline void zero(float* p, size_t n, hipStream_t s) {
    fzero_k<<<dim3((unsigned)((n + 255) / 256)), 256, 0, s>>>(p, n);
}
static inline void axpy(const float* gate, int E, int e, const float* src,
                        float* dst, int T, int O, hipStream_t s) {
    size_t total = (size_t)T * Bc * O;
    axpy_gate_k<<<dim3((unsigned)((total + 255) / 256)), 256, 0, s>>>(
        gate, E, e, src, dst, T, Bc, O);
}
// one expert: in[B,Din] (bf16) -> GEMM -> LN/affine/ReLU (->bf16) -> GEMM -> out fp32
static inline void expert(const unsigned short* inbf,
                          const float* w1, const float* b1,
                          const float* g, const float* be,
                          const float* w2, const float* b2,
                          int Din, int Hh, int Dout,
                          unsigned short* wbf, float* tmpH, unsigned short* tmpHbf,
                          float* out, hipStream_t s) {
    cvt(w1, wbf, (size_t)Din * Hh, s);
    gemm(inbf, wbf, b1, tmpH, Bc, Hh, Din, 0, s);
    ln_affine_relu_bf16_k<<<Bc, 256, 0, s>>>(tmpH, g, be, tmpHbf, Hh);
    cvt(w2, wbf, (size_t)Hh * Dout, s);
    gemm(tmpHbf, wbf, b2, out, Bc, Dout, Hh, 0, s);
}

extern "C" void kernel_launch(void* const* d_in, const int* in_sizes, int n_in,
                              void* d_out, int out_size, void* d_ws, size_t ws_size,
                              hipStream_t stream) {
    const float* x       = (const float*)d_in[0];
    const float* mmoe_w1 = (const float*)d_in[1];
    const float* mmoe_b1 = (const float*)d_in[2];
    const float* mmoe_g  = (const float*)d_in[3];
    const float* mmoe_be = (const float*)d_in[4];
    const float* mmoe_w2 = (const float*)d_in[5];
    const float* mmoe_b2 = (const float*)d_in[6];
    const float* mmoe_gw = (const float*)d_in[7];
    const float* mmoe_gb = (const float*)d_in[8];
    const float* c0_sw1  = (const float*)d_in[9];
    const float* c0_sb1  = (const float*)d_in[10];
    const float* c0_sg   = (const float*)d_in[11];
    const float* c0_sbe  = (const float*)d_in[12];
    const float* c0_sw2  = (const float*)d_in[13];
    const float* c0_sb2  = (const float*)d_in[14];
    const float* c0_tw1  = (const float*)d_in[15];
    const float* c0_tb1  = (const float*)d_in[16];
    const float* c0_tg   = (const float*)d_in[17];
    const float* c0_tbe  = (const float*)d_in[18];
    const float* c0_tw2  = (const float*)d_in[19];
    const float* c0_tb2  = (const float*)d_in[20];
    const float* c0_gw   = (const float*)d_in[21];
    const float* c0_gb   = (const float*)d_in[22];
    const float* c0_sgw  = (const float*)d_in[23];
    const float* c0_sgb  = (const float*)d_in[24];
    const float* c1_sw1  = (const float*)d_in[25];
    const float* c1_sb1  = (const float*)d_in[26];
    const float* c1_sg   = (const float*)d_in[27];
    const float* c1_sbe  = (const float*)d_in[28];
    const float* c1_sw2  = (const float*)d_in[29];
    const float* c1_sb2  = (const float*)d_in[30];
    const float* c1_tw1  = (const float*)d_in[31];
    const float* c1_tb1  = (const float*)d_in[32];
    const float* c1_tg   = (const float*)d_in[33];
    const float* c1_tbe  = (const float*)d_in[34];
    const float* c1_tw2  = (const float*)d_in[35];
    const float* c1_tb2  = (const float*)d_in[36];
    const float* c1_gw   = (const float*)d_in[37];
    const float* c1_gb   = (const float*)d_in[38];
    const float* tow_w1  = (const float*)d_in[39];
    const float* tow_b1  = (const float*)d_in[40];
    const float* tow_w2  = (const float*)d_in[41];
    const float* tow_b2  = (const float*)d_in[42];
    float* out = (float*)d_out;

    // ---- workspace carve-up (~315 MiB; reuse buffers across stages) ----
    char* ws = (char*)d_ws;
    size_t off = 0;
    auto carve = [&](size_t bytes) -> void* {
        void* p = ws + off;
        off += (bytes + 255) & ~(size_t)255;
        return p;
    };
    unsigned short* wbf    = (unsigned short*)carve((size_t)Dc * HMc * 2);
    float*          tmpH   = (float*)carve((size_t)Bc * HMc * 4);
    unsigned short* tmpHbf = (unsigned short*)carve((size_t)Bc * HMc * 2);
    float*          tmpO   = (float*)carve((size_t)Bc * OMc * 4);
    unsigned short* x_bf   = (unsigned short*)carve((size_t)Bc * Dc * 2);
    float*          task_hs    = (float*)carve((size_t)Tc * Bc * OMc * 4);
    unsigned short* task_hs_bf = (unsigned short*)carve((size_t)Tc * Bc * OMc * 2);
    float*          shared_h    = (float*)carve((size_t)Bc * OMc * 4);
    unsigned short* shared_h_bf = (unsigned short*)carve((size_t)Bc * OMc * 2);
    float*          task0    = (float*)carve((size_t)Tc * Bc * OCc * 4);
    unsigned short* task0_bf = (unsigned short*)carve((size_t)Tc * Bc * OCc * 2);
    float*          shared0    = (float*)carve((size_t)Bc * OCc * 4);
    unsigned short* shared0_bf = (unsigned short*)carve((size_t)Bc * OCc * 2);
    float*          task1    = (float*)carve((size_t)Tc * Bc * OCc * 4);
    unsigned short* task1_bf = (unsigned short*)carve((size_t)Tc * Bc * OCc * 2);
    float* gate_m = (float*)carve((size_t)Tc * Bc * Ec * 4);
    float* gate0  = (float*)carve((size_t)Tc * Bc * (Pc + Sc) * 4);
    float* sgate0 = (float*)carve((size_t)Bc * (Tc * Pc + Sc) * 4);
    float* gate1  = (float*)carve((size_t)Tc * Bc * (Pc + Sc) * 4);

    const int gateBlocks = (Tc * Bc + 7) / 8;   // 8 wave-rows per 256-thr block

    // ================= MMoE =================
    cvt(x, x_bf, (size_t)Bc * Dc, stream);
    gate_softmax_k<<<gateBlocks, 256, 0, stream>>>(
        x, 0, mmoe_gw, mmoe_gb, gate_m, Bc, Dc, Ec, Tc);
    zero(task_hs, (size_t)Tc * Bc * OMc, stream);
    for (int e = 0; e < Ec; ++e) {
        expert(x_bf,
               mmoe_w1 + (size_t)e * Dc * HMc, mmoe_b1 + (size_t)e * HMc,
               mmoe_g  + (size_t)e * HMc,      mmoe_be + (size_t)e * HMc,
               mmoe_w2 + (size_t)e * HMc * OMc, mmoe_b2 + (size_t)e * OMc,
               Dc, HMc, OMc, wbf, tmpH, tmpHbf, tmpO, stream);
        axpy(gate_m, Ec, e, tmpO, task_hs, Tc, OMc, stream);
    }
    mean3_k<<<(Bc * OMc + 255) / 256, 256, 0, stream>>>(
        task_hs, shared_h, (size_t)Bc * OMc, (size_t)Bc * OMc);
    cvt(task_hs, task_hs_bf, (size_t)Tc * Bc * OMc, stream);
    cvt(shared_h, shared_h_bf, (size_t)Bc * OMc, stream);

    // ================= CGC layer 0 =================
    gate_softmax_k<<<gateBlocks, 256, 0, stream>>>(
        task_hs, (size_t)Bc * OMc, c0_gw, c0_gb, gate0, Bc, OMc, Pc + Sc, Tc);
    gate_softmax_k<<<(Bc + 7) / 8, 256, 0, stream>>>(
        shared_h, 0, c0_sgw, c0_sgb, sgate0, Bc, OMc, Tc * Pc + Sc, 1);
    zero(task0, (size_t)Tc * Bc * OCc, stream);
    zero(shared0, (size_t)Bc * OCc, stream);
    for (int s = 0; s < Sc; ++s) {       // shared experts
        expert(shared_h_bf,
               c0_sw1 + (size_t)s * OMc * HCc, c0_sb1 + (size_t)s * HCc,
               c0_sg  + (size_t)s * HCc,       c0_sbe + (size_t)s * HCc,
               c0_sw2 + (size_t)s * HCc * OCc, c0_sb2 + (size_t)s * OCc,
               OMc, HCc, OCc, wbf, tmpH, tmpHbf, tmpO, stream);
        axpy(gate0, Pc + Sc, Pc + s, tmpO, task0, Tc, OCc, stream);
        axpy(sgate0, Tc * Pc + Sc, Tc * Pc + s, tmpO, shared0, 1, OCc, stream);
    }
    for (int t = 0; t < Tc; ++t)          // task-specific experts
        for (int p = 0; p < Pc; ++p) {
            const size_t tp = (size_t)t * Pc + p;
            expert(task_hs_bf + (size_t)t * Bc * OMc,
                   c0_tw1 + tp * OMc * HCc, c0_tb1 + tp * HCc,
                   c0_tg  + tp * HCc,       c0_tbe + tp * HCc,
                   c0_tw2 + tp * HCc * OCc, c0_tb2 + tp * OCc,
                   OMc, HCc, OCc, wbf, tmpH, tmpHbf, tmpO, stream);
            axpy(gate0 + (size_t)t * Bc * (Pc + Sc), Pc + Sc, p,
                 tmpO, task0 + (size_t)t * Bc * OCc, 1, OCc, stream);
            axpy(sgate0, Tc * Pc + Sc, t * Pc + p, tmpO, shared0, 1, OCc, stream);
        }
    cvt(task0, task0_bf, (size_t)Tc * Bc * OCc, stream);
    cvt(shared0, shared0_bf, (size_t)Bc * OCc, stream);

    // ================= CGC layer 1 =================
    gate_softmax_k<<<gateBlocks, 256, 0, stream>>>(
        task0, (size_t)Bc * OCc, c1_gw, c1_gb, gate1, Bc, OCc, Pc + Sc, Tc);
    zero(task1, (size_t)Tc * Bc * OCc, stream);
    for (int s = 0; s < Sc; ++s) {
        expert(shared0_bf,
               c1_sw1 + (size_t)s * OCc * HCc, c1_sb1 + (size_t)s * HCc,
               c1_sg  + (size_t)s * HCc,       c1_sbe + (size_t)s * HCc,
               c1_sw2 + (size_t)s * HCc * OCc, c1_sb2 + (size_t)s * OCc,
               OCc, HCc, OCc, wbf, tmpH, tmpHbf, tmpO, stream);
        axpy(gate1, Pc + Sc, Pc + s, tmpO, task1, Tc, OCc, stream);
    }
    for (int t = 0; t < Tc; ++t)
        for (int p = 0; p < Pc; ++p) {
            const size_t tp = (size_t)t * Pc + p;
            expert(task0_bf + (size_t)t * Bc * OCc,
                   c1_tw1 + tp * OCc * HCc, c1_tb1 + tp * HCc,
                   c1_tg  + tp * HCc,       c1_tbe + tp * HCc,
                   c1_tw2 + tp * HCc * OCc, c1_tb2 + tp * OCc,
                   OCc, HCc, OCc, wbf, tmpH, tmpHbf, tmpO, stream);
            axpy(gate1 + (size_t)t * Bc * (Pc + Sc), Pc + Sc, p,
                 tmpO, task1 + (size_t)t * Bc * OCc, 1, OCc, stream);
        }
    cvt(task1, task1_bf, (size_t)Tc * Bc * OCc, stream);

    // ================= task towers =================
    for (int t = 0; t < Tc; ++t) {
        cvt(tow_w1 + (size_t)t * OCc * THc, wbf, (size_t)OCc * THc, stream);
        gemm(task1_bf + (size_t)t * Bc * OCc, wbf, tow_b1 + (size_t)t * THc,
             tmpO, Bc, THc, OCc, /*relu=*/1, stream);
        tower_out_k<<<(Bc + 255) / 256, 256, 0, stream>>>(
            tmpO, tow_w2 + (size_t)t * THc, tow_b2 + t, out + (size_t)t * Bc,
            Bc, THc);
    }
}
